// RNN_54924041781709
// MI455X (gfx1250) — compile-verified, matrix operand order checked by reading, not run on hardware
//
#include <hip/hip_runtime.h>
#include <hip/hip_bf16.h>
#include <math.h>

#define VOCAB 10000
#define EMB   512
#define HID   1024
#define SEQ   128
#define BATCH 64

typedef __attribute__((ext_vector_type(16))) __bf16 v16bf;
typedef __attribute__((ext_vector_type(8)))  __bf16 v8bf;
typedef __attribute__((ext_vector_type(8)))  float  v8f;

union FragU { v16bf v; v8bf h[2]; };

// ---------------------------------------------------------------------------
// f32 -> bf16 conversion (weights, initial hidden state)
// ---------------------------------------------------------------------------
__global__ __launch_bounds__(256)
void f32_to_bf16_kernel(const float* __restrict__ src, __bf16* __restrict__ dst, int n) {
    int i = (blockIdx.x * 256 + threadIdx.x) * 4;
    if (i + 3 < n) {
        float4 v = *(const float4*)(src + i);
        dst[i + 0] = (__bf16)v.x;
        dst[i + 1] = (__bf16)v.y;
        dst[i + 2] = (__bf16)v.z;
        dst[i + 3] = (__bf16)v.w;
    } else {
        for (; i < n; ++i) dst[i] = (__bf16)src[i];
    }
}

// ---------------------------------------------------------------------------
// Embedding gather: xbf[s*B+b, :] = bf16(emb[inputs[s*B+b], :])
// ---------------------------------------------------------------------------
__global__ __launch_bounds__(128)
void embed_kernel(const int* __restrict__ inputs, const float* __restrict__ emb,
                  __bf16* __restrict__ xbf) {
    int sb  = blockIdx.x;
    int tok = inputs[sb];
    int e   = threadIdx.x * 4;
    float4 v = *(const float4*)(emb + (size_t)tok * EMB + e);
    __bf16* d = xbf + (size_t)sb * EMB + e;
    d[0] = (__bf16)v.x; d[1] = (__bf16)v.y; d[2] = (__bf16)v.z; d[3] = (__bf16)v.w;
}

// ---------------------------------------------------------------------------
// One RNN layer, one timestep: out = tanh(X @ Wx^T + H @ Wh^T + bias)
// X:(64,Kx)  H:(64,HID)  out:(64,HID). 256 tiles of 16x16, 8 waves/block,
// 32 blocks; fragments straight from L2 (weights are L2-resident).
// ---------------------------------------------------------------------------
__device__ __forceinline__ v16bf load_frag(const __bf16* __restrict__ base, int ld,
                                           int row0, int k0, int r, int koff) {
    FragU u;
    const __bf16* p = base + (size_t)(row0 + r) * ld + k0 + koff;
    u.h[0] = *(const v8bf*)(p);
    u.h[1] = *(const v8bf*)(p + 16);
    return u.v;
}

__global__ __launch_bounds__(256)
void rnn_step_kernel(const __bf16* __restrict__ X, int ldx,
                     const __bf16* __restrict__ Wx,
                     const __bf16* __restrict__ H,
                     const __bf16* __restrict__ Wh,
                     const float*  __restrict__ bias,
                     __bf16* __restrict__ out_state,
                     __bf16* __restrict__ out_seq,     // nullable
                     float*  __restrict__ out_f32) {   // nullable (final hidden)
    const int lane = threadIdx.x & 31;
    const int wave = threadIdx.x >> 5;
    const int tile = blockIdx.x * 8 + wave;      // 0..255
    const int tm   = (tile & 3) * 16;
    const int tn   = (tile >> 2) * 16;
    const int r    = lane & 15;
    const int koff = (lane >> 4) * 8;

    v8f acc = {};

    for (int k = 0; k < ldx; k += 32) {
        v16bf a = load_frag(X,  ldx, tm, k, r, koff);
        v16bf b = load_frag(Wx, ldx, tn, k, r, koff);
        acc = __builtin_amdgcn_wmma_f32_16x16x32_bf16(false, a, false, b,
                                                      (short)0, acc, false, false);
    }
    for (int k = 0; k < HID; k += 32) {
        v16bf a = load_frag(H,  HID, tm, k, r, koff);
        v16bf b = load_frag(Wh, HID, tn, k, r, koff);
        acc = __builtin_amdgcn_wmma_f32_16x16x32_bf16(false, a, false, b,
                                                      (short)0, acc, false, false);
    }

    const int   n     = tn + r;
    const float bn    = bias[n];
    const int   mbase = tm + (lane >> 4) * 8;
#pragma unroll
    for (int i = 0; i < 8; ++i) {
        float  v  = tanhf(acc[i] + bn);
        int    m  = mbase + i;
        __bf16 bv = (__bf16)v;
        out_state[(size_t)m * HID + n] = bv;
        if (out_seq) out_seq[(size_t)m * HID + n] = bv;
        if (out_f32) out_f32[(size_t)m * HID + n] = v;
    }
}

// ---------------------------------------------------------------------------
// Output projection: out(8192,10000) = H1(8192,1024) @ Wout^T + bout
// 128x128 macro tiles, K-chunks of 64, DOUBLE-BUFFERED LDS staged via
// CDNA5 async DMA (GLOBAL_LOAD_ASYNC_TO_LDS_B128, tracked by ASYNCcnt).
// 8 waves each own a 64x32 region (4x2 tiles -> 16 WMMAs between barriers).
// ---------------------------------------------------------------------------
#define BM 128
#define BN 128
#define BK 64
#define NCHUNK (HID / BK)                 // 16
#define BUF_ELEMS ((BM + BN) * BK)        // bf16 elems per buffer

// async memory->LDS DMA: per lane 16B; INST_OFFSET is added to BOTH the LDS
// dest addr and the global addr (ISA 08 §4.4), so one base pair covers 64B.
#define ASYNC_B128(ldsaddr, gaddr, OFF)                                      \
    asm volatile("global_load_async_to_lds_b128 %0, %1, off offset:" OFF     \
                 :: "v"(ldsaddr), "v"(gaddr) : "memory")

__device__ __forceinline__ void async_copy_64B(unsigned lds, unsigned long long g) {
    ASYNC_B128(lds, g, "0");
    ASYNC_B128(lds, g, "16");
    ASYNC_B128(lds, g, "32");
    ASYNC_B128(lds, g, "48");
}

__global__ __launch_bounds__(256)
void logits_kernel(const __bf16* __restrict__ A,     // (SEQ*BATCH, HID)
                   const __bf16* __restrict__ Wb,    // (VOCAB, HID)
                   const float*  __restrict__ bout,  // (VOCAB)
                   float* __restrict__ out) {        // (SEQ*BATCH, VOCAB)
    __shared__ __align__(16) __bf16 smem[2 * BUF_ELEMS];   // 64 KB of 320 KB LDS

    const int m0   = blockIdx.y * BM;
    const int n0   = blockIdx.x * BN;
    const int tid  = threadIdx.x;
    const int lane = tid & 31;
    const int wave = tid >> 5;
    const int wm   = (wave >> 2) * 64;   // 0 / 64
    const int wn   = (wave & 3) * 32;    // 0,32,64,96
    const int r    = lane & 15;
    const int koff = (lane >> 4) * 8;

    // staging: 2 threads per row, 32 contiguous bf16 (64B) each
    const int srow = tid >> 1;           // 0..127
    const int scol = (tid & 1) * 32;     // 0 / 32

    // global source pointers (clamp Wout row at the N edge: garbage only ever
    // lands in accumulator columns whose stores are guarded below)
    const __bf16* gA = A + (size_t)(m0 + srow) * HID + scol;
    int nrow = n0 + srow; if (nrow >= VOCAB) nrow = VOCAB - 1;
    const __bf16* gB = Wb + (size_t)nrow * HID + scol;

    // per-lane LDS byte addresses (flat LDS addr keeps offset in addr[31:0])
    const unsigned smemBase  = (unsigned)(uintptr_t)smem;
    const unsigned bufStride = (unsigned)(BUF_ELEMS * 2);
    const unsigned ldsA = smemBase + (unsigned)((srow * BK + scol) * 2);
    const unsigned ldsB = ldsA + (unsigned)(BM * BK * 2);

    v8f acc[4][2];
#pragma unroll
    for (int mi = 0; mi < 4; ++mi)
#pragma unroll
        for (int ni = 0; ni < 2; ++ni) acc[mi][ni] = (v8f){};

    // ---- prologue: DMA chunk 0 into buffer 0 (8 async b128 per thread) ----
    async_copy_64B(ldsA, (unsigned long long)(uintptr_t)gA);
    async_copy_64B(ldsB, (unsigned long long)(uintptr_t)gB);

    for (int c = 0; c < NCHUNK; ++c) {
        const int cur = c & 1;
        if (c + 1 < NCHUNK) {
            // DMA next chunk into the other buffer (freed by last iteration's
            // trailing barrier)
            const unsigned la = ldsA + (unsigned)(cur ^ 1) * bufStride;
            const unsigned lb = ldsB + (unsigned)(cur ^ 1) * bufStride;
            async_copy_64B(la, (unsigned long long)(uintptr_t)(gA + (c + 1) * BK));
            async_copy_64B(lb, (unsigned long long)(uintptr_t)(gB + (c + 1) * BK));
            // ASYNCcnt retires in order: <=8 outstanding means chunk c landed
            asm volatile("s_wait_asynccnt 0x8" ::: "memory");
        } else {
            asm volatile("s_wait_asynccnt 0x0" ::: "memory");
        }
        __syncthreads();   // all waves' chunk-c data visible in LDS

        const __bf16* sA = smem + cur * BUF_ELEMS;
        const __bf16* sB = sA + BM * BK;
#pragma unroll
        for (int ks = 0; ks < 2; ++ks) {       // two K=32 steps per chunk
            const int kb = ks * 32 + koff;
            FragU af[4], bf[2];
#pragma unroll
            for (int mi = 0; mi < 4; ++mi) {
                const __bf16* p = sA + (wm + mi * 16 + r) * BK + kb;
                af[mi].h[0] = *(const v8bf*)(p);
                af[mi].h[1] = *(const v8bf*)(p + 16);
            }
#pragma unroll
            for (int ni = 0; ni < 2; ++ni) {
                const __bf16* p = sB + (wn + ni * 16 + r) * BK + kb;
                bf[ni].h[0] = *(const v8bf*)(p);
                bf[ni].h[1] = *(const v8bf*)(p + 16);
            }
#pragma unroll
            for (int mi = 0; mi < 4; ++mi)
#pragma unroll
                for (int ni = 0; ni < 2; ++ni)
                    acc[mi][ni] = __builtin_amdgcn_wmma_f32_16x16x32_bf16(
                        false, af[mi].v, false, bf[ni].v, (short)0, acc[mi][ni],
                        false, false);
        }
        __syncthreads();   // buffer 'cur' free for the next iteration's DMA
    }

    // epilogue: +bias, f32 store (guard N edge; 10000 % 128 != 0)
#pragma unroll
    for (int ni = 0; ni < 2; ++ni) {
        int n = n0 + wn + ni * 16 + r;
        if (n < VOCAB) {
            float bn = bout[n];
#pragma unroll
            for (int mi = 0; mi < 4; ++mi) {
                int mb = m0 + wm + mi * 16 + (lane >> 4) * 8;
#pragma unroll
                for (int i = 0; i < 8; ++i)
                    out[(size_t)(mb + i) * VOCAB + n] = acc[mi][ni][i] + bn;
            }
        }
    }
}

// ---------------------------------------------------------------------------
// host launcher
// ---------------------------------------------------------------------------
static inline int cvt_grid(int n) { return (n / 4 + 255) / 256; }

extern "C" void kernel_launch(void* const* d_in, const int* in_sizes, int n_in,
                              void* d_out, int out_size, void* d_ws, size_t ws_size,
                              hipStream_t stream) {
    const int*   inputs = (const int*)  d_in[0];
    const float* hidden = (const float*)d_in[1];
    const float* emb    = (const float*)d_in[2];
    const float* Wx0    = (const float*)d_in[3];
    const float* Wh0    = (const float*)d_in[4];
    const float* bh0    = (const float*)d_in[5];
    const float* Wx1    = (const float*)d_in[6];
    const float* Wh1    = (const float*)d_in[7];
    const float* bh1    = (const float*)d_in[8];
    const float* Wout   = (const float*)d_in[9];
    const float* bout   = (const float*)d_in[10];
    float* out = (float*)d_out;
    float* hidden_final = out + (size_t)SEQ * BATCH * VOCAB;  // (2, B, HID)

    // ---- workspace carve (all bf16 buffers) ----
    char*  ws  = (char*)d_ws;
    size_t off = 0;
    auto carve = [&](size_t bytes) -> void* {
        off = (off + 255) & ~(size_t)255;
        void* p = ws + off;
        off += bytes;
        return p;
    };
    __bf16* xbf    = (__bf16*)carve((size_t)SEQ * BATCH * EMB * 2);
    __bf16* h1_all = (__bf16*)carve((size_t)SEQ * BATCH * HID * 2);
    __bf16* h0s[2] = { (__bf16*)carve((size_t)BATCH * HID * 2),
                       (__bf16*)carve((size_t)BATCH * HID * 2) };
    __bf16* h1s[2] = { (__bf16*)carve((size_t)BATCH * HID * 2),
                       (__bf16*)carve((size_t)BATCH * HID * 2) };
    __bf16* Wx0b = (__bf16*)carve((size_t)HID * EMB * 2);
    __bf16* Wh0b = (__bf16*)carve((size_t)HID * HID * 2);
    __bf16* Wx1b = (__bf16*)carve((size_t)HID * HID * 2);
    __bf16* Wh1b = (__bf16*)carve((size_t)HID * HID * 2);
    __bf16* Woutb = (__bf16*)carve((size_t)VOCAB * HID * 2);
    (void)ws_size; (void)n_in; (void)in_sizes; (void)out_size;

    // ---- weight / state conversions ----
    f32_to_bf16_kernel<<<cvt_grid(HID * EMB), 256, 0, stream>>>(Wx0, Wx0b, HID * EMB);
    f32_to_bf16_kernel<<<cvt_grid(HID * HID), 256, 0, stream>>>(Wh0, Wh0b, HID * HID);
    f32_to_bf16_kernel<<<cvt_grid(HID * HID), 256, 0, stream>>>(Wx1, Wx1b, HID * HID);
    f32_to_bf16_kernel<<<cvt_grid(HID * HID), 256, 0, stream>>>(Wh1, Wh1b, HID * HID);
    f32_to_bf16_kernel<<<cvt_grid(VOCAB * HID), 256, 0, stream>>>(Wout, Woutb, VOCAB * HID);
    f32_to_bf16_kernel<<<cvt_grid(BATCH * HID), 256, 0, stream>>>(hidden, h0s[0], BATCH * HID);
    f32_to_bf16_kernel<<<cvt_grid(BATCH * HID), 256, 0, stream>>>(hidden + (size_t)BATCH * HID,
                                                                  h1s[0], BATCH * HID);

    // ---- embedding gather ----
    embed_kernel<<<SEQ * BATCH, 128, 0, stream>>>(inputs, emb, xbf);

    // ---- recurrent phase: 2 small WMMA GEMMs per timestep ----
    for (int t = 0; t < SEQ; ++t) {
        int rd = t & 1, wr = (t + 1) & 1;
        bool last = (t == SEQ - 1);
        rnn_step_kernel<<<32, 256, 0, stream>>>(
            xbf + (size_t)t * BATCH * EMB, EMB, Wx0b,
            h0s[rd], Wh0b, bh0,
            h0s[wr], (__bf16*)nullptr,
            last ? hidden_final : (float*)nullptr);
        rnn_step_kernel<<<32, 256, 0, stream>>>(
            h0s[wr], HID, Wx1b,
            h1s[rd], Wh1b, bh1,
            h1s[wr], h1_all + (size_t)t * BATCH * HID,
            last ? hidden_final + (size_t)BATCH * HID : (float*)nullptr);
    }

    // ---- bulk output projection: 8192 x 10000 x 1024 ----
    dim3 grid((VOCAB + BN - 1) / BN, (SEQ * BATCH) / BM);
    logits_kernel<<<grid, 256, 0, stream>>>(h1_all, Woutb, bout, out);
}
